// GSGM_37495064494296
// MI455X (gfx1250) — matrix operand order repaired and checked by reading, not compiled, not run on hardware
//
#include <hip/hip_runtime.h>

typedef __attribute__((ext_vector_type(16))) _Float16 v16h;
typedef __attribute__((ext_vector_type(8)))  _Float16 v8h;
typedef __attribute__((ext_vector_type(2)))  _Float16 v2h;
typedef __attribute__((ext_vector_type(8)))  float    v8f;

#define NSTEPS 1000

// Compiler-ordering fence only: intra-wave LDS ops are processed in order by
// the DS pipeline (CDNA5 ISA: "LDS ... In order"), so no hardware barrier is
// needed for same-wave store->load staging; we only stop compiler reordering.
#define LDS_FENCE() asm volatile("" ::: "memory")

struct GsgmParams {
  const float* reco_w;  const float* reco_b;
  const float* mix_w;   const float* mix_b;
  const float* embed_w; const float* embed_b;
  const float* inp_w;   const float* inp_b;
  const float* out_w;   const float* out_b;
  const float* res_w[4]; const float* res_b[4];
  const float* d1_w[4];  const float* d1_b[4];
  const float* d2_w[4];  const float* d2_b[4];
};

__device__ __forceinline__ float leaky(float x) { return fmaxf(x, 0.01f * x); }

__device__ __forceinline__ v8f wmma_f16(v16h a, v16h b, v8f c) {
  // emits v_wmma_f32_16x16x32_f16
  return __builtin_amdgcn_wmma_f32_16x16x32_f16(false, a, false, b, (short)0, c,
                                                false, false);
}

__device__ __forceinline__ v8f bcast8(float b) {
  v8f r;
#pragma unroll
  for (int j = 0; j < 8; j++) r[j] = b;
  return r;
}

// Load a 16x32 f16 A-fragment for this lane from a row-major [16][32] LDS tile
// (tile holds *buffer positions*; staged-consumer weight B-frags are permuted
// to match). Lane: row m = lane%16; positions {kb..kb+7} U {kb+16..kb+23}.
__device__ __forceinline__ v16h load_afrag(const _Float16* buf, int m, int kb) {
  v8h lo = *(const v8h*)(buf + m * 32 + kb);
  v8h hh = *(const v8h*)(buf + m * 32 + kb + 16);
  v16h r;
#pragma unroll
  for (int j = 0; j < 8; j++) { r[j] = lo[j]; r[j + 8] = hh[j]; }
  return r;
}

__global__ __launch_bounds__(32) void gsgm_sampler_kernel(
    const float* __restrict__ cond, const float* __restrict__ x_init,
    float* __restrict__ out, GsgmParams P) {
  // Weight B-fragments: 0,1 embed(n0,n1); 2,3 inp; 4+6b+{0,1}=res,{2,3}=d1,{4,5}=d2; 28 out
  __shared__ __align__(32) v16h wfrag[29][32];
  __shared__ __align__(16) _Float16 sZ[16 * 32];
  __shared__ __align__(16) _Float16 sU[16 * 32];
  __shared__ __align__(16) _Float16 sX[16 * 8];   // x staging, cols 6..7 stay zero
  __shared__ float sRecoMix[16][16];              // act-free mix of reco + b_mix
  __shared__ float sTvec[16];

  const int lane = (int)threadIdx.x;  // single wave32 per block
  const int hi = lane >> 4;
  const int nn = lane & 15;
  const int kbA = hi * 8;
  const int rowBase = (int)blockIdx.x * 16;  // 32 blocks x 16 rows = 512

  // ---- init: weight B-fragments (B layout: n=lane%16, k_phys=8*hi+h (h<8), 16+8*hi+h-8 (h>=8)). -----
  // Fragments 4..28 consume STAGED activations whose buffer position of
  // logical feature c is pos(c)=2*(c%16)+(c/16); their k rows take the inverse
  // permutation. Fragments 0..3 (embed/inp) use identity k, and carry their
  // bias vector in a spare K row (embed: k==16, inp: k==6) multiplied by a
  // constant-1.0 A slot. Built through LDS (rolled loop, compact code), then
  // hoisted into registers for the whole 1000-step loop.
  for (int f = 0; f < 29; f++) {
    const float* mat; const float* bvec = nullptr;
    int K, ld, ncols, nofs, biasRow = -1;
    bool permK = (f >= 4);
    if (f < 2) {
      mat = P.embed_w; K = 16; ld = 32; ncols = 32; nofs = (f & 1) * 16;
      bvec = P.embed_b; biasRow = 16;
    } else if (f < 4) {
      mat = P.inp_w; K = 6; ld = 32; ncols = 32; nofs = (f & 1) * 16;
      bvec = P.inp_b; biasRow = 6;
    } else if (f == 28) {
      mat = P.out_w; K = 32; ld = 6; ncols = 6; nofs = 0;
    } else {
      int b = (f - 4) / 6, r = (f - 4) % 6;
      mat = (r < 2) ? P.res_w[b] : (r < 4) ? P.d1_w[b] : P.d2_w[b];
      K = 32; ld = 32; ncols = 32; nofs = (r & 1) * 16;
    }
    int n = nofs + nn;
    v16h v;
#pragma unroll
    for (int h = 0; h < 16; h++) {
      int kp = (h < 8) ? (hi * 8 + h) : (16 + hi * 8 + (h - 8));
      int k = permK ? ((kp & 1) * 16 + (kp >> 1)) : kp;
      float val = 0.f;
      if (k < K && n < ncols) val = mat[k * ld + n];
      else if (k == biasRow && bvec != nullptr) val = bvec[n];
      v[h] = (_Float16)val;
    }
    wfrag[f][lane] = v;
  }
  for (int e = lane; e < 128; e += 32) sX[e] = (_Float16)0.f;
  // step-invariant: recoMix[r][k] = b_mix[k] + sum_j act(reco)_j * mix_w[16+j][k]
  for (int e = lane; e < 256; e += 32) {
    int r = e >> 4, k = e & 15;
    int grow = rowBase + r;
    float acc = P.mix_b[k];
    for (int j = 0; j < 16; j++) {
      float rj = P.reco_b[j];
      for (int i = 0; i < 6; i++) rj += cond[grow * 6 + i] * P.reco_w[i * 16 + j];
      acc += leaky(rj) * P.mix_w[(16 + j) * 16 + k];
    }
    sRecoMix[r][k] = acc;
  }
  __syncthreads();

  // ---- hoist all 29 B-fragments into registers for the whole loop ---------
  // (wave32 has up to 1024 VGPRs; with one wave per SIMD the high-VGPR window
  //  is free real estate, and this removes all per-step weight loads.)
  v16h Wf[29];
#pragma unroll
  for (int f = 0; f < 29; f++) Wf[f] = wfrag[f][lane];

  // ---- loop-invariant register state --------------------------------------
  float bd0[4], bd1[4], bs0[4], bs1[4];
#pragma unroll
  for (int b = 0; b < 4; b++) {
    bd0[b] = P.d1_b[b][nn];  bd1[b] = P.d1_b[b][16 + nn];
    bs0[b] = P.res_b[b][nn]  + P.d2_b[b][nn];
    bs1[b] = P.res_b[b][16 + nn] + P.d2_b[b][16 + nn];
  }
  const float bo = (nn < 6) ? P.out_b[nn] : 0.f;
  // this lane's recoMix slice (identity-k slots of cA)
  float rm[8];
#pragma unroll
  for (int h = 0; h < 8; h++) rm[h] = sRecoMix[nn][kbA + h];
  // mix_w time-branch column for the 16 tvec lanes
  float wmcol[16];
#pragma unroll
  for (int i = 0; i < 16; i++) wmcol[i] = (lane < 16) ? P.mix_w[i * 16 + lane] : 0.f;

  // x in WMMA C-layout: lane holds column n=nn (<6), rows m=j+8*hi
  v8f xr, meanr;
#pragma unroll
  for (int j = 0; j < 8; j++) { xr[j] = 0.f; meanr[j] = 0.f; }
  if (nn < 6) {
#pragma unroll
    for (int j = 0; j < 8; j++) xr[j] = x_init[(rowBase + j + 8 * hi) * 6 + nn];
  }

  const float Bc = 4.5399929762e-05f;  // atan(e^{-10})
  const float Aa = 1.5707055269f;      // atan(e^{10}) - atan(e^{-10})
  // Fourier frequencies * 1000:  1000 * 10^(-4i/7), i = 0..7  (inline literals)
  const float kF1000[8] = {1000.0f,    268.26958f, 71.968567f,  19.306977f,
                           5.1794747f, 1.3894955f, 0.37275937f, 0.1f};

#pragma unroll 1
  for (int s = 0; s < NSTEPS; s++) {
    float t = (float)(NSTEPS - s) * (1.f / NSTEPS);
    // time branch of mix: tvec[k] = sum_i temb_i * mix_w[i][k] (16 lanes)
    if (lane < 16) {
      float acc = 0.f;
#pragma unroll
      for (int i = 0; i < 8; i++) {
        float ang = t * kF1000[i];
        acc += __sinf(ang) * wmcol[i] + __cosf(ang) * wmcol[8 + i];
      }
      sTvec[lane] = acc;
    }
    LDS_FENCE();
    // sigmoid(-2 log tan th) == cos^2(th)  => alpha=cos, sigma=sin
    float th = Aa * t + Bc;
    float th2 = Aa * (t - 1.f / NSTEPS) + Bc;
    float alpha = __cosf(th), sigma = __sinf(th);
    float alpha2 = __cosf(th2), sigma2 = __sinf(th2);

    // c = act(recoMix + tvec) as A-fragment; slot for k==16 carries 1.0 so the
    // embed fragment's bias row is applied inside the WMMA.
    v16h ca;
#pragma unroll
    for (int h = 0; h < 16; h++) ca[h] = (_Float16)0.f;
#pragma unroll
    for (int h = 0; h < 8; h++) {
      float val = leaky(rm[h] + sTvec[kbA + h]);
      ca[h] = (_Float16)val;
    }
    if (!hi) ca[8] = (_Float16)1.0f;  // k_phys==16 lives in slot 8 of lanes 0..15
    v8f zero8 = bcast8(0.f);
    v8f e0 = wmma_f16(ca, Wf[0], zero8);
    v8f e1 = wmma_f16(ca, Wf[1], zero8);
#pragma unroll
    for (int j = 0; j < 8; j++) { e0[j] = leaky(e0[j]); e1[j] = leaky(e1[j]); }

    // w1 = x @ W_inp + b_inp + e   (bias via spare k==6 row; e as WMMA C)
    if (nn < 6) {
#pragma unroll
      for (int j = 0; j < 8; j++) sX[(j + 8 * hi) * 8 + nn] = (_Float16)xr[j];
    }
    LDS_FENCE();
    v16h xa;
#pragma unroll
    for (int h = 0; h < 16; h++) xa[h] = (_Float16)0.f;
    if (!hi) {
      v8h lo = *(const v8h*)(sX + nn * 8);
#pragma unroll
      for (int j = 0; j < 8; j++) xa[j] = lo[j];
      xa[6] = (_Float16)1.0f;  // k==6 -> inp bias row
    }
    v8f h0 = wmma_f16(xa, Wf[2], e0);
    v8f h1 = wmma_f16(xa, Wf[3], e1);

    // blocks: maintain w_i = h_i + e (e folded into the res-chain C for
    // bq<3); block 3 accumulates onto plain bsum so the head sees h.
#pragma unroll
    for (int bq = 0; bq < 4; bq++) {
      const int fb = 4 + 6 * bq;
#pragma unroll
      for (int j = 0; j < 8; j++) {
        int m2 = j + 8 * hi;
        v2h p;
        p[0] = (_Float16)h0[j];
        p[1] = (_Float16)h1[j];
        *(v2h*)(sZ + m2 * 32 + 2 * nn) = p;
      }
      LDS_FENCE();
      v16h za = load_afrag(sZ, nn, kbA);
      v8f u0 = wmma_f16(za, Wf[fb + 2], bcast8(bd0[bq]));
      v8f u1 = wmma_f16(za, Wf[fb + 3], bcast8(bd1[bq]));
#pragma unroll
      for (int j = 0; j < 8; j++) {
        int m2 = j + 8 * hi;
        v2h p;
        p[0] = (_Float16)leaky(u0[j]);
        p[1] = (_Float16)leaky(u1[j]);
        *(v2h*)(sU + m2 * 32 + 2 * nn) = p;
      }
      LDS_FENCE();
      v16h ua = load_afrag(sU, nn, kbA);
      v8f c0, c1;
      if (bq < 3) {
#pragma unroll
        for (int j = 0; j < 8; j++) { c0[j] = e0[j] + bs0[bq]; c1[j] = e1[j] + bs1[bq]; }
      } else {
        c0 = bcast8(bs0[3]);
        c1 = bcast8(bs1[3]);
      }
      c0 = wmma_f16(za, Wf[fb + 0], c0);  // z @ W_res + ...
      c0 = wmma_f16(ua, Wf[fb + 4], c0);  // + u @ W_d2
      c1 = wmma_f16(za, Wf[fb + 1], c1);
      c1 = wmma_f16(ua, Wf[fb + 5], c1);
      h0 = c0; h1 = c1;
    }

    // v = h @ W_out + b_out  (N padded 6->16; k rows permuted like staging)
#pragma unroll
    for (int j = 0; j < 8; j++) {
      int m2 = j + 8 * hi;
      v2h p;
      p[0] = (_Float16)h0[j];
      p[1] = (_Float16)h1[j];
      *(v2h*)(sZ + m2 * 32 + 2 * nn) = p;
    }
    LDS_FENCE();
    v16h ha = load_afrag(sZ, nn, kbA);
    v8f vv = wmma_f16(ha, Wf[28], bcast8(bo));

    // elementwise sampler update (all in C-layout registers)
#pragma unroll
    for (int j = 0; j < 8; j++) {
      float mn = alpha * xr[j] - sigma * vv[j];
      float ep = sigma * xr[j] + alpha * vv[j];
      meanr[j] = mn;
      xr[j] = alpha2 * mn + sigma2 * ep;
    }
    LDS_FENCE();
  }

  if (nn < 6) {
#pragma unroll
    for (int j = 0; j < 8; j++) out[(rowBase + j + 8 * hi) * 6 + nn] = meanr[j];
  }
}

extern "C" void kernel_launch(void* const* d_in, const int* in_sizes, int n_in,
                              void* d_out, int out_size, void* d_ws, size_t ws_size,
                              hipStream_t stream) {
  (void)d_ws; (void)ws_size; (void)out_size;
  const float* cond = (const float*)d_in[0];
  const float* x_init = (const float*)d_in[1];
  GsgmParams P;
  auto F = [&](int i) { return (const float*)d_in[i]; };
  if (n_in >= 3 && in_sizes[2] == 96) {
    // recursive insertion-order flatten: reco,mix,embed,inp,out,blocks[{res,d1,d2}], each {w,b}
    int i = 2;
    P.reco_w = F(i++);  P.reco_b = F(i++);
    P.mix_w = F(i++);   P.mix_b = F(i++);
    P.embed_w = F(i++); P.embed_b = F(i++);
    P.inp_w = F(i++);   P.inp_b = F(i++);
    P.out_w = F(i++);   P.out_b = F(i++);
    for (int b = 0; b < 4; b++) {
      P.res_w[b] = F(i++); P.res_b[b] = F(i++);
      P.d1_w[b] = F(i++);  P.d1_b[b] = F(i++);
      P.d2_w[b] = F(i++);  P.d2_b[b] = F(i++);
    }
  } else {
    // JAX pytree flatten (sorted dict keys): blocks[{d1,d2,res}],embed,inp,mix,out,reco; each {b,w}
    int i = 2;
    for (int b = 0; b < 4; b++) {
      P.d1_b[b] = F(i++);  P.d1_w[b] = F(i++);
      P.d2_b[b] = F(i++);  P.d2_w[b] = F(i++);
      P.res_b[b] = F(i++); P.res_w[b] = F(i++);
    }
    P.embed_b = F(i++); P.embed_w = F(i++);
    P.inp_b = F(i++);   P.inp_w = F(i++);
    P.mix_b = F(i++);   P.mix_w = F(i++);
    P.out_b = F(i++);   P.out_w = F(i++);
    P.reco_b = F(i++);  P.reco_w = F(i++);
  }
  // 512 independent rows -> 32 blocks x 1 wave32, 16-row WMMA tile per wave.
  gsgm_sampler_kernel<<<32, 32, 0, stream>>>(cond, x_init, (float*)d_out, P);
}
